// IndependentStyleGAN2GeneratorBackbone_14559939133779
// MI455X (gfx1250) — matrix operand order repaired and checked by reading, not compiled, not run on hardware
//
#include <hip/hip_runtime.h>

// ---------------------------------------------------------------------------
// StyleGAN2-style generator backbone for MI455X (gfx1250, wave32, WMMA+TDM).
//
// Activations are PIXEL-MAJOR [b][p][c] so that
//   Y[p][co] = sum_ci X[p][ci] * Wm[co][ci]
// maps to WMMA with no transposes:
//   A (M=pixel,K=cin) fragment: contiguous K per lane  -> LDS b128 loads
//   B (K=cin,N=cout) = Wm^T:   element(k,n)=Wm[n][k]   -> contiguous rows of
//       Wm, loaded DIRECTLY from global (weights are L1/L2-resident, shared
//       by all waves -> broadcast hits; no LDS, no transpose).
// The A tile (128 px x 32 cin bf16 = 8KB) is DMA'd into LDS by the Tensor
// Data Mover (tensor_load_to_lds), double-buffered with s_wait_tensorcnt so
// the next tile's DMA overlaps the current tile's WMMAs.
// ---------------------------------------------------------------------------

typedef __bf16 bf16x8  __attribute__((ext_vector_type(8)));
typedef __bf16 bf16x16 __attribute__((ext_vector_type(16)));
typedef float  f32x8   __attribute__((ext_vector_type(8)));
typedef unsigned int u32x4 __attribute__((ext_vector_type(4)));
typedef int          i32x4 __attribute__((ext_vector_type(4)));
typedef int          i32x8 __attribute__((ext_vector_type(8)));

#define HW   16384   // 128*128 pixels
#define NB   4       // batch
#define COND 512
#define C0   512

#if __has_builtin(__builtin_amdgcn_tensor_load_to_lds) && \
    __has_builtin(__builtin_amdgcn_s_wait_tensorcnt)
#define USE_TDM 1
#else
#define USE_TDM 0
#endif

// ---------------------------------------------------------------------------
// Small GEMV: out[b*N+n] = vec[b]·mat[:,n] (+bias). mat stored [K,N].
// ---------------------------------------------------------------------------
__global__ __launch_bounds__(256)
void matvec_kernel(const float* __restrict__ mat, const float* __restrict__ vec,
                   const float* __restrict__ bias, float* __restrict__ out,
                   int N, int K, int total) {
  int idx = blockIdx.x * blockDim.x + threadIdx.x;
  if (idx >= total) return;
  int b = idx / N, n = idx % N;
  const float* v = vec + b * K;
  float s = bias ? bias[n] : 0.0f;
  for (int k = 0; k < K; ++k) s += v[k] * mat[k * N + n];
  out[idx] = s;
}

// ---------------------------------------------------------------------------
// Weight demodulation: wm[b,o,i] = w[o,i]*style[b,i] * rsqrt(sum_i(.)^2+eps)
// One wave32 per (b,o) row; butterfly reduction across 32 lanes.
// ---------------------------------------------------------------------------
__global__ __launch_bounds__(128)
void modweight_kernel(const float* __restrict__ w, const float* __restrict__ style,
                      __bf16* __restrict__ wm, int M, int K) {
  int wave = blockIdx.x * (blockDim.x >> 5) + (threadIdx.x >> 5);
  int lane = threadIdx.x & 31;
  int b = wave / M, o = wave % M;
  const float* wrow = w + (size_t)o * K;
  const float* srow = style + (size_t)b * K;
  float ss = 0.0f;
  for (int i = lane; i < K; i += 32) {
    float v = wrow[i] * srow[i];
    ss += v * v;
  }
  #pragma unroll
  for (int d = 16; d >= 1; d >>= 1) ss += __shfl_xor(ss, d, 32);
  float dem = rsqrtf(ss + 1e-8f);
  __bf16* orow = wm + ((size_t)b * M + o) * K;
  for (int i = lane; i < K; i += 32)
    orow[i] = (__bf16)(wrow[i] * srow[i] * dem);
}

// ---------------------------------------------------------------------------
// First layer: x0[b,p,c] = sin(coord0*P[b,0,c] + coord1*P[b,1,c] + bias[b,c])
// Pixel-major output; c coalesced, coords are wave-uniform scalar loads.
// ---------------------------------------------------------------------------
__global__ __launch_bounds__(256)
void first_layer_kernel(const float* __restrict__ coords, const float* __restrict__ proj,
                        const float* __restrict__ bias0, __bf16* __restrict__ out) {
  int c = blockIdx.x * 256 + threadIdx.x;
  int p = blockIdx.y;
  int b = blockIdx.z;
  float c_0 = coords[((size_t)b * 2 + 0) * HW + p];
  float c_1 = coords[((size_t)b * 2 + 1) * HW + p];
  float v = c_0 * proj[b * 1024 + c] + c_1 * proj[b * 1024 + 512 + c] + bias0[b * 512 + c];
  out[((size_t)b * HW + p) * C0 + c] = (__bf16)__sinf(v);
}

// ---------------------------------------------------------------------------
// TDM: DMA a 2-D tile (32 contiguous cin elems x 128 pixel rows, row stride
// K elements, 2-byte data) from global into LDS. D# per CDNA5 ISA ch.8.
// This toolchain exposes the 6-arg builtin: (g0, g1, g2, g3, ext, cpol);
// groups 2/3 + extension words are zero for a plain 2-D tile.
// ---------------------------------------------------------------------------
#if USE_TDM
__device__ __forceinline__ void tdm_load_tile(const __bf16* gsrc, unsigned lds_off, int K) {
  unsigned long long ga = (unsigned long long)(size_t)gsrc;
  u32x4 g0;
  g0[0] = 1u;                                   // count=1 (valid), user mode
  g0[1] = lds_off;                              // lds_addr (bytes)
  g0[2] = (unsigned)ga;                         // global_addr[31:0]
  g0[3] = (unsigned)((ga >> 32) & 0x01FFFFFFu)  // global_addr[56:32]
        | 0x80000000u;                          // type=2 ("image")
  i32x8 g1;
  g1[0] = 0x10000;                              // data_size=1 -> 2 bytes
  g1[1] = (int)((unsigned)K << 16);             // tensor_dim0[15:0]
  g1[2] = (int)(((unsigned)K >> 16) | ((unsigned)HW << 16)); // dim0 hi | dim1 lo
  g1[3] = (int)(((unsigned)HW >> 16) | (32u << 16));         // dim1 hi | tile_dim0=32
  g1[4] = 128;                                  // tile_dim1=128, tile_dim2=0
  g1[5] = K;                                    // tensor_dim0_stride[31:0]
  g1[6] = 0;                                    // stride hi | dim1_stride lo
  g1[7] = 0;
  i32x4 z4 = {0, 0, 0, 0};                      // groups 2/3 unused (2-D tile)
  i32x8 z8 = {0, 0, 0, 0, 0, 0, 0, 0};          // extension words unused
  __builtin_amdgcn_tensor_load_to_lds(g0, g1, z4, z4, z8, 0);
}
#endif

// ---------------------------------------------------------------------------
// bf16 GEMM: Y[b,p,n] = sum_k X[b,p,k] * Wm[b,n,k] + bias[n]  (all n=cout)
// 256 threads = 8 waves. Block tile: 128 pixels x 64 couts.
// Wave w owns pixels [w*16, w*16+16); computes 4 successive 16x16 tiles
// along cout. A fragment from LDS (TDM-staged); B fragments directly from
// global Wm rows (contiguous, cache-broadcast across waves).
// ISA fragment layouts (wave32):
//   A 16x32 bf16: lane<16 -> K{0..7,16..23}; lane>=16 -> K{8..15,24..31}
//   B 32x16 bf16: col=lane&15; lane<16 -> K{0..7,16..23}; lane>=16 -> K{8..15,24..31}
//   C/D f32:      vgpr r -> row r (lanes 0-15) / r+8 (lanes 16-31)
// ---------------------------------------------------------------------------
__global__ __launch_bounds__(256)
void gemm_bf16_kernel(const __bf16* __restrict__ Wm, const __bf16* __restrict__ X,
                      const float* __restrict__ bias, __bf16* __restrict__ Y,
                      int M, int K) {
  __shared__ __attribute__((aligned(16))) __bf16 ldsA[2][128 * 32];

  const int t    = threadIdx.x;
  const int b    = blockIdx.z;
  const int p0   = blockIdx.x * 128;   // pixel tile
  const int n0   = blockIdx.y * 64;    // cout tile
  const int wave = t >> 5, lane = t & 31;
  const int lm   = lane & 15, hi = lane >> 4;

  const __bf16* Wb = Wm + (size_t)b * M * K;
  const __bf16* Xb = X + ((size_t)b * HW + p0) * K;  // tile base, row stride K

  f32x8 acc[4] = {};
  const int nk = K >> 5;

#if USE_TDM
  const unsigned lds0 = (unsigned)(size_t)&ldsA[0][0];
  const unsigned lds1 = (unsigned)(size_t)&ldsA[1][0];
  if (wave == 0) tdm_load_tile(Xb, lds0, K);
#endif

  for (int i = 0; i < nk; ++i) {
    const int kt = i << 5;
#if USE_TDM
    if (wave == 0) {
      if (i + 1 < nk) {
        tdm_load_tile(Xb + kt + 32, (i & 1) ? lds0 : lds1, K);
        __builtin_amdgcn_s_wait_tensorcnt(1);  // older DMA (current buf) done
      } else {
        __builtin_amdgcn_s_wait_tensorcnt(0);  // last tile: drain
      }
    }
    const __bf16* A = ldsA[i & 1];
#else
    {  // cooperative synchronous staging: 2x b128 per thread, no transpose
      const __bf16* src = Xb + (size_t)(t >> 1) * K + kt + (t & 1) * 16;
      *(uint4*)&ldsA[0][(t >> 1) * 32 + (t & 1) * 16]     = *(const uint4*)src;
      *(uint4*)&ldsA[0][(t >> 1) * 32 + (t & 1) * 16 + 8] = *(const uint4*)(src + 8);
    }
    const __bf16* A = ldsA[0];
#endif
    __syncthreads();

    bf16x8 a_lo = *(const bf16x8*)&A[(wave * 16 + lm) * 32 + hi * 8];
    bf16x8 a_hi = *(const bf16x8*)&A[(wave * 16 + lm) * 32 + 16 + hi * 8];
    bf16x16 af = __builtin_shufflevector(a_lo, a_hi,
        0, 1, 2, 3, 4, 5, 6, 7, 8, 9, 10, 11, 12, 13, 14, 15);

    #pragma unroll
    for (int s = 0; s < 4; ++s) {
      const __bf16* wrow = Wb + (size_t)(n0 + s * 16 + lm) * K + kt + hi * 8;
      bf16x8 b_lo = *(const bf16x8*)(wrow);
      bf16x8 b_hi = *(const bf16x8*)(wrow + 16);
      bf16x16 bf = __builtin_shufflevector(b_lo, b_hi,
          0, 1, 2, 3, 4, 5, 6, 7, 8, 9, 10, 11, 12, 13, 14, 15);
      acc[s] = __builtin_amdgcn_wmma_f32_16x16x32_bf16(
          false, af, false, bf, (short)0, acc[s], false, false);
    }
    __syncthreads();
  }

  // Epilogue: fused conv bias (indexed by cout = lane column), bf16 store
  #pragma unroll
  for (int s = 0; s < 4; ++s) {
    int n = n0 + s * 16 + lm;
    float bn = bias[n];
    #pragma unroll
    for (int r = 0; r < 8; ++r) {
      int p = p0 + wave * 16 + hi * 8 + r;
      Y[((size_t)b * HW + p) * M + n] = (__bf16)(acc[s][r] + bn);
    }
  }
}

// ---------------------------------------------------------------------------
// RGB projection: out[b,j,p] (+)= sum_c rgb_w[j,c]*X[b,p,c] + rgb_b[j]
// Pixel-major X: per-thread contiguous vector loads.
// ---------------------------------------------------------------------------
__global__ __launch_bounds__(256)
void rgb_kernel(const __bf16* __restrict__ X, const float* __restrict__ rgb_w,
                const float* __restrict__ rgb_b, float* __restrict__ out,
                int C, int add) {
  int idx = blockIdx.x * 256 + threadIdx.x;   // b*HW + p
  int b = idx / HW, p = idx % HW;
  const __bf16* xr = X + ((size_t)b * HW + p) * C;
  float s0 = 0.f, s1 = 0.f, s2 = 0.f;
  for (int c = 0; c < C; c += 8) {
    bf16x8 xv = *(const bf16x8*)(xr + c);
    #pragma unroll
    for (int i = 0; i < 8; ++i) {
      float x = (float)xv[i];
      s0 += rgb_w[c + i] * x;
      s1 += rgb_w[C + c + i] * x;
      s2 += rgb_w[2 * C + c + i] * x;
    }
  }
  float* o = out + (size_t)b * 3 * HW;
  s0 += rgb_b[0]; s1 += rgb_b[1]; s2 += rgb_b[2];
  if (add) { o[p] += s0; o[HW + p] += s1; o[2 * HW + p] += s2; }
  else     { o[p]  = s0; o[HW + p]  = s1; o[2 * HW + p]  = s2; }
}

// ---------------------------------------------------------------------------
// Launcher. Input order (depth-first over setup_inputs dict):
//  0 coordinates [4,2,128,128]   1 condition [4,512]
//  2 proj_w [512,1024]  3 proj_b  4 bias_w [512,512]  5 bias_b
//  per block k (base 6+8k): cond_w0, cond_w1, conv_w0, conv_w1,
//                           conv_b0, conv_b1, rgb_w, rgb_b
// ---------------------------------------------------------------------------
extern "C" void kernel_launch(void* const* d_in, const int* in_sizes, int n_in,
                              void* d_out, int out_size, void* d_ws, size_t ws_size,
                              hipStream_t stream) {
  (void)in_sizes; (void)n_in; (void)out_size; (void)ws_size;
  const float* coords = (const float*)d_in[0];
  const float* cond   = (const float*)d_in[1];
  const float* proj_w = (const float*)d_in[2];
  const float* proj_b = (const float*)d_in[3];
  const float* bias_w = (const float*)d_in[4];
  const float* bias_b = (const float*)d_in[5];
  float* out = (float*)d_out;

  char* ws = (char*)d_ws;
  float*  proj   = (float*)(ws);                         // 4*1024 f32
  float*  bias0  = (float*)(ws + 16384);                 // 4*512  f32
  float*  style0 = (float*)(ws + 24576);                 // 6 slots of 4*512 f32
  __bf16* wm     = (__bf16*)(ws + 73728);                // <= 4*512*512 bf16 (2MB)
  __bf16* bufA   = (__bf16*)(ws + 2170880);              // 4*HW*512 bf16 (64MB)
  __bf16* bufB   = (__bf16*)(ws + 2170880 + 67108864);   // 4*HW*512 bf16 (64MB)

  // --- condition-dependent vectors (fp32) ---
  matvec_kernel<<<(NB * 1024 + 255) / 256, 256, 0, stream>>>(
      proj_w, cond, proj_b, proj, 1024, COND, NB * 1024);
  matvec_kernel<<<(NB * 512 + 255) / 256, 256, 0, stream>>>(
      bias_w, cond, bias_b, bias0, 512, COND, NB * 512);

  const int cins[6] = {512, 512, 512, 512, 256, 256};
  for (int l = 0; l < 6; ++l) {
    int blk = l / 2, li = l % 2;
    const float* cw = (const float*)d_in[6 + 8 * blk + li];
    float* st = style0 + l * (NB * 512);
    matvec_kernel<<<(NB * cins[l] + 255) / 256, 256, 0, stream>>>(
        cw, cond, nullptr, st, cins[l], COND, NB * cins[l]);
  }

  // --- sinusoidal first layer -> bf16 pixel-major activations in bufA ---
  first_layer_kernel<<<dim3(C0 / 256, HW, NB), 256, 0, stream>>>(
      coords, proj, bias0, bufA);

  // --- demod-conv = modweight + WMMA GEMM ---
  auto conv = [&](int blk, int li, int layer, const __bf16* Xb, __bf16* Yb,
                  int M, int K) {
    const float* w  = (const float*)d_in[6 + 8 * blk + 2 + li];
    const float* cb = (const float*)d_in[6 + 8 * blk + 4 + li];
    const float* st = style0 + layer * (NB * 512);
    modweight_kernel<<<M, 128, 0, stream>>>(w, st, wm, M, K);  // NB*M waves
    gemm_bf16_kernel<<<dim3(HW / 128, M / 64, NB), 256, 0, stream>>>(
        wm, Xb, cb, Yb, M, K);
  };

  // Block 0: 512->512->512, rgb from C=512
  conv(0, 0, 0, bufA, bufB, 512, 512);
  conv(0, 1, 1, bufB, bufA, 512, 512);
  rgb_kernel<<<NB * HW / 256, 256, 0, stream>>>(
      bufA, (const float*)d_in[12], (const float*)d_in[13], out, 512, 0);

  // Block 1: 512->512->256, rgb from C=256
  conv(1, 0, 2, bufA, bufB, 512, 512);
  conv(1, 1, 3, bufB, bufA, 256, 512);
  rgb_kernel<<<NB * HW / 256, 256, 0, stream>>>(
      bufA, (const float*)d_in[20], (const float*)d_in[21], out, 256, 1);

  // Block 2: 256->256->128, rgb from C=128
  conv(2, 0, 4, bufA, bufB, 256, 256);
  conv(2, 1, 5, bufB, bufA, 128, 256);
  rgb_kernel<<<NB * HW / 256, 256, 0, stream>>>(
      bufA, (const float*)d_in[28], (const float*)d_in[29], out, 128, 1);
}
